// PrunableLeViTAttention_33947421507744
// MI455X (gfx1250) — matrix operand order repaired and checked by reading, not compiled, not run on hardware
//
#include <hip/hip_runtime.h>
#include <hip/hip_bf16.h>

typedef __attribute__((ext_vector_type(16))) _Float16 v16h;
typedef __attribute__((ext_vector_type(8)))  _Float16 v8h;
typedef __attribute__((ext_vector_type(4)))  _Float16 v4h;
typedef __attribute__((ext_vector_type(8)))  float    v8f;
typedef __attribute__((ext_vector_type(4)))  float    v4f;

#define RES        14
#define N_TOK      196
#define DIM        384
#define KEY_DIM    32
#define NUM_HEADS  12
#define HEAD_QKV   192      // 32 q + 32 k + 128 v per head
#define VAL_DIM    128
#define QKV_DIM    2304
#define VAL_ATTN   1536
#define BATCH      128
#define MTOT       (BATCH * N_TOK)   // 25088
#define NPAD       224               // 196 padded to multiple of 32
#define NBLK       13                // row blocks needed (rows 0..207 cover 196)

static __device__ __forceinline__ v16h mk16(v8h lo, v8h hi) {
  v16h r;
#pragma unroll
  for (int i = 0; i < 8; ++i) { r[i] = lo[i]; r[8 + i] = hi[i]; }
  return r;
}

// ---------------------------------------------------------------------------
// Fold BN (and optionally the q attention scale) into f16 weights + f32 bias.
// ---------------------------------------------------------------------------
__global__ void prep_w_kernel(const float* __restrict__ w,
                              const float* __restrict__ gamma,
                              const float* __restrict__ beta,
                              const float* __restrict__ mean,
                              const float* __restrict__ var,
                              _Float16* __restrict__ w_h,
                              float* __restrict__ bias,
                              int rows, int cols, int fold_q_scale) {
  int idx = blockIdx.x * blockDim.x + threadIdx.x;
  if (idx >= rows * cols) return;
  int o = idx / cols;
  float inv = gamma[o] * rsqrtf(var[o] + 1e-5f);
  float s = 1.0f;
  if (fold_q_scale && (o % HEAD_QKV) < KEY_DIM) s = 0.17677669529663687f; // 32^-0.5
  w_h[idx] = (_Float16)(w[idx] * inv * s);
  if ((idx % cols) == 0) bias[o] = (beta[o] - mean[o] * inv) * s;
}

// Expand relative-position bias to dense [NUM_HEADS][N_TOK][NPAD] (pad cols 0).
__global__ void prep_bias_kernel(const float* __restrict__ attn_biases,
                                 float* __restrict__ bias_full) {
  int idx = blockIdx.x * blockDim.x + threadIdx.x;
  if (idx >= NUM_HEADS * N_TOK * NPAD) return;
  int h = idx / (N_TOK * NPAD);
  int rem = idx - h * (N_TOK * NPAD);
  int r = rem / NPAD;
  int c = rem - r * NPAD;
  float v = 0.f;
  if (c < N_TOK) {
    int dy = r / RES - c / RES; dy = dy < 0 ? -dy : dy;
    int dx = r % RES - c % RES; dx = dx < 0 ? -dx : dx;
    v = attn_biases[h * N_TOK + dy * RES + dx];
  }
  bias_full[idx] = v;
}

__global__ void cvt_h_kernel(const float* __restrict__ src,
                             _Float16* __restrict__ dst, int n4) {
  int i = blockIdx.x * blockDim.x + threadIdx.x;
  if (i >= n4) return;
  v4f s = ((const v4f*)src)[i];
  v4h d;
#pragma unroll
  for (int j = 0; j < 4; ++j) d[j] = (_Float16)s[j];
  ((v4h*)dst)[i] = d;
}

// ---------------------------------------------------------------------------
// WMMA GEMM:  C[M x N] = A[M x K] * W[N x K]^T + bias[N]
// 8 waves / block, each wave computes a 32x64 strip (2 A-frags x 4 B-frags,
// 8 WMMAs per K-step) -> block tile 256x64. Fragment loads for the next
// K-step are issued before the WMMAs of the current step (software pipeline;
// last step re-loads the same fragments so no OOB access).
// ---------------------------------------------------------------------------
template <int STORE_F16>
__global__ void gemm_bn_kernel(const _Float16* __restrict__ A,
                               const _Float16* __restrict__ W,
                               const float* __restrict__ bias,
                               _Float16* __restrict__ outH,
                               float* __restrict__ outF,
                               int K, int ldc) {
  const int wave  = threadIdx.x >> 5;
  const int lane  = threadIdx.x & 31;
  const int row0  = blockIdx.x * 256 + wave * 32;
  const int col0  = blockIdx.y * 64;
  const int lrow  = lane & 15;
  const int koffA = (lane & 16) ? 8 : 0;    // A: K half-block select
  const int koffB = (lane & 16) ? 8 : 0;    // B: K half-block select

  const _Float16* pa0 = A + (size_t)(row0 + lrow) * K + koffA;
  const _Float16* pa1 = pa0 + (size_t)16 * K;
  const _Float16* pw[4];
#pragma unroll
  for (int t = 0; t < 4; ++t)
    pw[t] = W + (size_t)(col0 + t * 16 + lrow) * K + koffB;

  v8f acc[2][4] = {};
  v16h a0 = mk16(*(const v8h*)pa0, *(const v8h*)(pa0 + 16));
  v16h a1 = mk16(*(const v8h*)pa1, *(const v8h*)(pa1 + 16));
  v16h b[4];
#pragma unroll
  for (int t = 0; t < 4; ++t)
    b[t] = mk16(*(const v8h*)pw[t], *(const v8h*)(pw[t] + 16));

  for (int k0 = 0; k0 < K; k0 += 32) {
    int kn = (k0 + 32 < K) ? k0 + 32 : k0;   // clamp: last step reloads
    v16h na0 = mk16(*(const v8h*)(pa0 + kn), *(const v8h*)(pa0 + kn + 16));
    v16h na1 = mk16(*(const v8h*)(pa1 + kn), *(const v8h*)(pa1 + kn + 16));
    v16h nb[4];
#pragma unroll
    for (int t = 0; t < 4; ++t)
      nb[t] = mk16(*(const v8h*)(pw[t] + kn), *(const v8h*)(pw[t] + kn + 16));
#pragma unroll
    for (int t = 0; t < 4; ++t) {
      acc[0][t] = __builtin_amdgcn_wmma_f32_16x16x32_f16(
          false, a0, false, b[t], (short)0, acc[0][t], false, false);
      acc[1][t] = __builtin_amdgcn_wmma_f32_16x16x32_f16(
          false, a1, false, b[t], (short)0, acc[1][t], false, false);
    }
    a0 = na0; a1 = na1;
#pragma unroll
    for (int t = 0; t < 4; ++t) b[t] = nb[t];
  }

  const int rbase = row0 + ((lane & 16) >> 1);  // lanes 16-31 -> rows +8
#pragma unroll
  for (int u = 0; u < 2; ++u) {
#pragma unroll
    for (int t = 0; t < 4; ++t) {
      int c = col0 + t * 16 + lrow;
      float bi = bias[c];
#pragma unroll
      for (int j = 0; j < 8; ++j) {
        size_t r = (size_t)(rbase + u * 16 + j);
        float v = acc[u][t][j] + bi;
        if (STORE_F16) outH[r * ldc + c] = (_Float16)v;
        else           outF[r * ldc + c] = v;
      }
    }
  }
}

// ---------------------------------------------------------------------------
// Fused attention: one workgroup (8 waves) per (batch, head).
// LDS: scores[224x224]f16 (base 0), q[224x32]h, k[224x32]h, v^T[128x224]h
// Total 186,368 B. Row-block outer / column-tile inner with the B fragment
// software-pipelined one tile ahead so WMMA + epilogue overlap the next load.
// ---------------------------------------------------------------------------
__global__ void attn_kernel(const _Float16* __restrict__ qkv,
                            const float* __restrict__ bias_full,
                            _Float16* __restrict__ outS) {
  extern __shared__ char smem[];
  _Float16* sS  = (_Float16*)smem;                 // NPAD x NPAD (f16)
  _Float16* sQ  = sS + NPAD * NPAD;                // NPAD x 32
  _Float16* sK  = sQ + NPAD * KEY_DIM;             // NPAD x 32
  _Float16* sVT = sK + NPAD * KEY_DIM;             // 128 x NPAD (transposed)

  const int b = blockIdx.x / NUM_HEADS;
  const int h = blockIdx.x % NUM_HEADS;
  const _Float16* base = qkv + (size_t)b * N_TOK * QKV_DIM + h * HEAD_QKV;
  const int tid = threadIdx.x;

  // stage q/k (vectorized, zero padded rows): 4 v8h per row of 32
  for (int idx = tid; idx < NPAD * KEY_DIM / 8; idx += 256) {
    int n = idx >> 2, d = (idx & 3) * 8;
    v8h q = {}, k = {};
    if (n < N_TOK) {
      q = *(const v8h*)(base + (size_t)n * QKV_DIM + d);
      k = *(const v8h*)(base + (size_t)n * QKV_DIM + KEY_DIM + d);
    }
    *(v8h*)(sQ + n * KEY_DIM + d) = q;
    *(v8h*)(sK + n * KEY_DIM + d) = k;
  }
  // stage v transposed: vector global load, scalar LDS scatter
  for (int idx = tid; idx < NPAD * VAL_DIM / 8; idx += 256) {
    int n = idx >> 4, c0 = (idx & 15) * 8;
    v8h v = {};
    if (n < N_TOK) v = *(const v8h*)(base + (size_t)n * QKV_DIM + 2 * KEY_DIM + c0);
#pragma unroll
    for (int i = 0; i < 8; ++i) sVT[(c0 + i) * NPAD + n] = v[i];
  }
  __syncthreads();

  const int wave  = tid >> 5, lane = tid & 31;
  const int lrow  = lane & 15;
  const int koffA = (lane & 16) ? 8 : 0;
  const int koffB = (lane & 16) ? 8 : 0;
  const int rofs  = (lane & 16) >> 1;

  // ---- raw scores = q @ k^T (scale pre-folded into q weights) ----
  const _Float16* pk = sK + lrow * KEY_DIM + koffB;   // + bj*16*KEY_DIM
  for (int bi = wave; bi < NBLK; bi += 8) {
    int row0 = bi * 16;
    const _Float16* pa = sQ + (row0 + lrow) * KEY_DIM + koffA;
    v16h a = mk16(*(const v8h*)pa, *(const v8h*)(pa + 16));
    _Float16* srow = sS + (row0 + rofs) * NPAD + lrow;
    v16h bf = mk16(*(const v8h*)pk, *(const v8h*)(pk + 16));
#pragma unroll 2
    for (int bj = 0; bj < 14; ++bj) {
      v16h cur = bf;
      int bn = (bj + 1 < 14) ? bj + 1 : bj;   // clamp: last tile reloads
      const _Float16* pn = pk + bn * 16 * KEY_DIM;
      bf = mk16(*(const v8h*)pn, *(const v8h*)(pn + 16));
      v8f acc = {};
      acc = __builtin_amdgcn_wmma_f32_16x16x32_f16(
          false, a, false, cur, (short)0, acc, false, false);
      _Float16* out = srow + bj * 16;
#pragma unroll
      for (int j = 0; j < 8; ++j) out[j * NPAD] = (_Float16)acc[j];
    }
  }
  __syncthreads();

  // ---- bias + row softmax (f16 in place, f32 math); zero pad columns ----
  for (int r = tid; r < N_TOK; r += 256) {
    _Float16* row = sS + r * NPAD;
    const float* gb = bias_full + ((size_t)h * N_TOK + r) * NPAD;
    float m = -1e30f;
    for (int c = 0; c < N_TOK; ++c) {
      float s = (float)row[c] + gb[c];
      row[c] = (_Float16)s;
      m = fmaxf(m, s);
    }
    float sum = 0.f;
    for (int c = 0; c < N_TOK; ++c) {
      float e = __expf((float)row[c] - m);
      row[c] = (_Float16)e;
      sum += e;
    }
    float rs = 1.f / sum;
    for (int c = 0; c < N_TOK; ++c) row[c] = (_Float16)((float)row[c] * rs);
    for (int c = N_TOK; c < NPAD; ++c) row[c] = (_Float16)0.f;
  }
  __syncthreads();

  // ---- out = P @ V, fused SiLU, store f16 ----
  // All 7 P fragments of a row block kept in registers (56 VGPRs), reused
  // across the 8 output column tiles; V fragment pipelined one K-step ahead.
  for (int bi = wave; bi < NBLK; bi += 8) {
    int row0 = bi * 16;
    v16h a[7];
#pragma unroll
    for (int kk = 0; kk < 7; ++kk) {
      const _Float16* ps = sS + (row0 + lrow) * NPAD + kk * 32 + koffA;
      a[kk] = mk16(*(const v8h*)ps, *(const v8h*)(ps + 16));
    }
#pragma unroll 2
    for (int bj = 0; bj < 8; ++bj) {
      int col0 = bj * 16;
      const _Float16* pvb = sVT + (col0 + lrow) * NPAD + koffB;
      v16h bf = mk16(*(const v8h*)pvb, *(const v8h*)(pvb + 16));
      v8f acc = {};
#pragma unroll
      for (int kk = 0; kk < 7; ++kk) {
        v16h cur = bf;
        int kn = (kk + 1 < 7) ? kk + 1 : kk;   // clamp: last step reloads
        const _Float16* pn = pvb + kn * 32;
        bf = mk16(*(const v8h*)pn, *(const v8h*)(pn + 16));
        acc = __builtin_amdgcn_wmma_f32_16x16x32_f16(
            false, a[kk], false, cur, (short)0, acc, false, false);
      }
      int c = col0 + lrow;
#pragma unroll
      for (int j = 0; j < 8; ++j) {
        int r = row0 + rofs + j;
        if (r < N_TOK) {
          float o = acc[j];
          o = o / (1.f + __expf(-o));   // silu
          outS[((size_t)b * N_TOK + r) * VAL_ATTN + h * VAL_DIM + c] = (_Float16)o;
        }
      }
    }
  }
}

// ---------------------------------------------------------------------------
extern "C" void kernel_launch(void* const* d_in, const int* in_sizes, int n_in,
                              void* d_out, int out_size, void* d_ws, size_t ws_size,
                              hipStream_t stream) {
  (void)in_sizes; (void)n_in; (void)out_size; (void)ws_size;
  const float* x          = (const float*)d_in[0];
  const float* qkv_w      = (const float*)d_in[1];
  const float* qkv_gamma  = (const float*)d_in[2];
  const float* qkv_beta   = (const float*)d_in[3];
  const float* qkv_mean   = (const float*)d_in[4];
  const float* qkv_var    = (const float*)d_in[5];
  const float* attn_bias  = (const float*)d_in[6];
  const float* proj_w     = (const float*)d_in[7];
  const float* proj_gamma = (const float*)d_in[8];
  const float* proj_beta  = (const float*)d_in[9];
  const float* proj_mean  = (const float*)d_in[10];
  const float* proj_var   = (const float*)d_in[11];

  char* ws = (char*)d_ws;
  size_t off = 0;
  auto carve = [&](size_t bytes) -> char* {
    char* p = ws + off;
    off += (bytes + 255) & ~(size_t)255;
    return p;
  };
  _Float16* x_h     = (_Float16*)carve((size_t)MTOT * DIM * 2);        // 19.3 MB
  _Float16* qkvw_h  = (_Float16*)carve((size_t)QKV_DIM * DIM * 2);     // 1.8 MB
  float*    qkv_b   = (float*)   carve((size_t)QKV_DIM * 4);
  _Float16* projw_h = (_Float16*)carve((size_t)DIM * VAL_ATTN * 2);    // 1.2 MB
  float*    proj_b  = (float*)   carve((size_t)DIM * 4);
  float*    bias_f  = (float*)   carve((size_t)NUM_HEADS * N_TOK * NPAD * 4); // 2.1 MB
  _Float16* qkv_h   = (_Float16*)carve((size_t)MTOT * QKV_DIM * 2);    // 115.6 MB
  _Float16* s_h     = (_Float16*)carve((size_t)MTOT * VAL_ATTN * 2);   // 77.1 MB

  cvt_h_kernel<<<(MTOT * DIM / 4 + 255) / 256, 256, 0, stream>>>(x, x_h, MTOT * DIM / 4);
  prep_w_kernel<<<(QKV_DIM * DIM + 255) / 256, 256, 0, stream>>>(
      qkv_w, qkv_gamma, qkv_beta, qkv_mean, qkv_var, qkvw_h, qkv_b,
      QKV_DIM, DIM, /*fold_q_scale=*/1);
  prep_w_kernel<<<(DIM * VAL_ATTN + 255) / 256, 256, 0, stream>>>(
      proj_w, proj_gamma, proj_beta, proj_mean, proj_var, projw_h, proj_b,
      DIM, VAL_ATTN, /*fold_q_scale=*/0);
  prep_bias_kernel<<<(NUM_HEADS * N_TOK * NPAD + 255) / 256, 256, 0, stream>>>(
      attn_bias, bias_f);

  // QKV projection: 25088 x 384 -> 25088 x 2304
  dim3 g1(MTOT / 256, QKV_DIM / 64);
  gemm_bn_kernel<1><<<g1, 256, 0, stream>>>(x_h, qkvw_h, qkv_b,
                                            qkv_h, nullptr, DIM, QKV_DIM);

  // fused attention per (batch, head)
  size_t smem = (size_t)NPAD * NPAD * 2          // sS (f16)
              + (size_t)NPAD * KEY_DIM * 2 * 2   // sQ + sK
              + (size_t)VAL_DIM * NPAD * 2;      // sVT => 186,368 B
  attn_kernel<<<BATCH * NUM_HEADS, 256, smem, stream>>>(qkv_h, bias_f, s_h);

  // output projection: 25088 x 1536 -> 25088 x 384 (f32 to d_out)
  dim3 g2(MTOT / 256, DIM / 64);
  gemm_bn_kernel<0><<<g2, 256, 0, stream>>>(s_h, projw_h, proj_b,
                                            nullptr, (float*)d_out,
                                            VAL_ATTN, DIM);
}
